// Linear_25116968747507
// MI455X (gfx1250) — matrix operand order repaired and checked by reading, not compiled, not run on hardware
//
#include <hip/hip_runtime.h>

// ---------------------------------------------------------------------------
// Fused SAE forward for MI455X (gfx1250, wave32):
//   pre_act = relu((x - b_dec) @ W_enc^T + b_enc)   [N=4096, L=32768], D=2048
//   top-32 per row  ->  out = b_dec + sum_k act_k * W_dec[idx_k]
// v4: bf16-W path stages W with CDNA5 async global->LDS copies (4 LDS
// buffers, 2-slice lookahead, s_wait_asynccnt) and now uses a SPLIT barrier:
// A-fragment ds_loads are issued before s_wait_asynccnt and s_barrier_signal,
// so they overlap the wait and barrier latency. Encoder: v_wmma_f32_16x16x32_bf16,
// m2 x n4 register tile per wave. fp32-W register-staged fallback if ws_size
// is too small for the bf16 W copy.
// ---------------------------------------------------------------------------

typedef __attribute__((ext_vector_type(16))) __bf16 v16bf;
typedef __attribute__((ext_vector_type(8)))  __bf16 v8bf;
typedef __attribute__((ext_vector_type(8)))  float  v8f;
typedef __attribute__((ext_vector_type(4)))  float  v4f;

constexpr int D_DIM    = 2048;            // reduction dim (== b_dec length)
constexpr int M_WG     = 32;              // tokens per workgroup
constexpr int N_CHUNK  = 512;             // latents per outer chunk (8 waves x 64)
constexpr int K_SLICE  = 32;              // k columns staged per slice
constexpr int N_SLICES = D_DIM / K_SLICE; // 64
constexpr int TOPK     = 32;

constexpr int W_STRIDE = K_SLICE + 8;     // 40 bf16/row: conflict-free b128 reads
constexpr int C_STRIDE = N_CHUNK + 1;     // 513 floats: conflict-free row scans
constexpr int T_STRIDE = 33;              // top-k list stride (bank spread)

constexpr unsigned SM_A_BYTES  = (unsigned)M_WG * D_DIM * 2;         // 131072
constexpr unsigned SM_W_OFF    = SM_A_BYTES;
constexpr unsigned WBUF_BYTES  = (unsigned)N_CHUNK * W_STRIDE * 2;   // 40960
constexpr unsigned SM_TV_BYTES = (unsigned)M_WG * T_STRIDE * 4;      // 4224

// fp32-W fallback: 2 register-staged buffers
constexpr unsigned F_W_BYTES = 2u * WBUF_BYTES;                      // 81920
constexpr unsigned F_TV_OFF  = SM_W_OFF + F_W_BYTES;
constexpr unsigned F_TI_OFF  = F_TV_OFF + SM_TV_BYTES;
constexpr unsigned F_TOTAL   = F_TI_OFF + SM_TV_BYTES;               // 221440

// bf16-W async path: 4 buffers, 2-slice lookahead
constexpr unsigned B_W_BYTES = 4u * WBUF_BYTES;                      // 163840
constexpr unsigned B_TV_OFF  = SM_W_OFF + B_W_BYTES;
constexpr unsigned B_TI_OFF  = B_TV_OFF + SM_TV_BYTES;
constexpr unsigned B_TOTAL   = B_TI_OFF + SM_TV_BYTES;               // 303360

__device__ __forceinline__ v8bf cvt8(v4f a, v4f b) {
    v8bf r;
    r[0] = (__bf16)a[0]; r[1] = (__bf16)a[1]; r[2] = (__bf16)a[2]; r[3] = (__bf16)a[3];
    r[4] = (__bf16)b[0]; r[5] = (__bf16)b[1]; r[6] = (__bf16)b[2]; r[7] = (__bf16)b[3];
    return r;
}

// ---- shared phase helpers --------------------------------------------------

__device__ __forceinline__ void init_topk(float* smTV, int* smTI, int tid) {
    if (tid < M_WG) {
        for (int j = 0; j < TOPK; ++j) {
            smTV[tid * T_STRIDE + j] = 0.0f;   // relu acts >= 0; zeros never win
            smTI[tid * T_STRIDE + j] = -1;
        }
    }
}

// A = (x - b_dec) -> bf16 LDS, XOR-swizzled 16B blocks (row stride 4096B is
// bank-aligned, so block (col/8) ^ (row&15) makes A-frag ds_load_b128
// conflict-free).
__device__ __forceinline__ void load_A(__bf16* smA, const float* x,
                                       const float* b_dec, int rowBase, int tid) {
    const int col = tid * 8;                    // 256 threads * 8 = 2048 cols
    v4f bd0 = *(const v4f*)(b_dec + col);
    v4f bd1 = *(const v4f*)(b_dec + col + 4);
    for (int r = 0; r < M_WG; ++r) {
        const float* xs = x + (size_t)(rowBase + r) * D_DIM + col;
        v4f a0 = *(const v4f*)(xs);
        v4f a1 = *(const v4f*)(xs + 4);
        a0 -= bd0; a1 -= bd1;
        const int blk = tid ^ (r & 15);         // (col>>3) == tid
        *(v8bf*)(smA + r * D_DIM + (blk << 3)) = cvt8(a0, a1);
    }
}

// relu(acc+bias) -> C-stage, then owner-thread sorted top-32 insertion.
__device__ __forceinline__ void chunk_epilogue(
    v8f (&acc)[2][4], const float (&bias)[4], float* smC, float* smTV, int* smTI,
    int tid, int wave, int lane, int n0) {
    const int l15  = lane & 15;
    const int crow = (lane >> 4) << 3;
#pragma unroll
    for (int m = 0; m < 2; ++m)
#pragma unroll
        for (int nt = 0; nt < 4; ++nt) {
            const int ccol = wave * 64 + nt * 16 + l15;
#pragma unroll
            for (int r = 0; r < 8; ++r)
                smC[(m * 16 + crow + r) * C_STRIDE + ccol] =
                    fmaxf(acc[m][nt][r] + bias[nt], 0.0f);
        }
    __syncthreads();
    if (tid < M_WG) {
        float* tV = smTV + tid * T_STRIDE;
        int*   tI = smTI + tid * T_STRIDE;
        const float* cr = smC + tid * C_STRIDE;
        for (int j = 0; j < N_CHUNK; ++j) {
            const float v = cr[j];
            if (v > tV[0]) {                     // running 32nd-max threshold
                int p = 1;
                while (p < TOPK && tV[p] < v) {
                    tV[p - 1] = tV[p]; tI[p - 1] = tI[p]; ++p;
                }
                tV[p - 1] = v; tI[p - 1] = n0 + j;
            }
        }
    }
    __syncthreads();
}

// Fused sparse decode: coalesced float4 gathers of W_dec rows (L2-resident).
__device__ __forceinline__ void decode(const float* smTV, const int* smTI,
                                       const float* W_dec, const float* b_dec,
                                       float* out, int rowBase, int wave, int lane) {
    for (int r = wave; r < M_WG; r += 8) {
        const float* tV = smTV + r * T_STRIDE;
        const int*   tI = smTI + r * T_STRIDE;
        float* orow = out + (size_t)(rowBase + r) * D_DIM;
        for (int d0 = 0; d0 < D_DIM; d0 += 128) {
            const int d = d0 + lane * 4;
            v4f acc4 = *(const v4f*)(b_dec + d);
#pragma unroll 4
            for (int j = 0; j < TOPK; ++j) {
                const int idx = tI[j];           // uniform across lanes
                if (idx >= 0) {
                    const float v = tV[j];
                    const v4f w = *(const v4f*)(W_dec + (size_t)idx * D_DIM + d);
                    acc4 += w * v;
                }
            }
            *(v4f*)(orow + d) = acc4;
        }
    }
}

// ---- W_enc fp32 -> bf16 conversion pre-pass --------------------------------

__global__ __launch_bounds__(256) void convert_w_bf16(
    const float* __restrict__ W, __bf16* __restrict__ Wb, long long n) {
    const long long stride = (long long)gridDim.x * 256 * 8;
    for (long long e = ((long long)blockIdx.x * 256 + threadIdx.x) * 8; e < n; e += stride) {
        v4f f0 = *(const v4f*)(W + e);
        v4f f1 = *(const v4f*)(W + e + 4);
        *(v8bf*)(Wb + e) = cvt8(f0, f1);
    }
}

// ---- bf16-W kernel: async global->LDS staging, 4 buffers, lookahead 2 ------

__device__ __forceinline__ void issue_slice_async(
    const __bf16* Wbf, unsigned ldsbuf, int n0, int s, int tid) {
    const int lat4 = tid >> 2;                  // 0..63
    const int c    = tid & 3;                   // 16B chunk within 32-elem row
#pragma unroll
    for (int p = 0; p < 8; ++p) {
        const int lat = p * 64 + lat4;
        const __bf16* g = Wbf + (size_t)(n0 + lat) * D_DIM + s * K_SLICE + c * 8;
        const unsigned l = ldsbuf + (unsigned)((lat * W_STRIDE + c * 8) * 2);
        asm volatile("global_load_async_to_lds_b128 %0, %1, off"
                     :: "v"(l), "v"(g) : "memory");
    }
}

__global__ __launch_bounds__(256) void sae_topk_fused_bf16w(
    const float* __restrict__ x, const __bf16* __restrict__ Wbf,
    const float* __restrict__ b_enc, const float* __restrict__ W_dec,
    const float* __restrict__ b_dec, float* __restrict__ out, int L)
{
    __shared__ __align__(16) unsigned char smem[B_TOTAL];
    __bf16* smA  = (__bf16*)(smem);
    __bf16* smW  = (__bf16*)(smem + SM_W_OFF);
    float*  smC  = (float*) (smem + SM_W_OFF);      // aliased over W buffers 0/1
    float*  smTV = (float*) (smem + B_TV_OFF);
    int*    smTI = (int*)   (smem + B_TI_OFF);

    const int tid  = threadIdx.x;
    const int lane = tid & 31;
    const int wave = tid >> 5;
    const int rowBase = blockIdx.x * M_WG;
    const unsigned wbase = (unsigned)(uintptr_t)smW;   // low 32 bits = LDS offset

    init_topk(smTV, smTI, tid);
    load_A(smA, x, b_dec, rowBase, tid);
    __syncthreads();

    const int kOffA = (lane >> 4) << 3;
    const int kOffB = (lane >> 4) << 3;
    const int l15   = lane & 15;

    for (int n0 = 0; n0 < L; n0 += N_CHUNK) {
        v8f acc[2][4];
#pragma unroll
        for (int m = 0; m < 2; ++m)
#pragma unroll
            for (int nt = 0; nt < 4; ++nt)
                acc[m][nt] = (v8f){0.f,0.f,0.f,0.f,0.f,0.f,0.f,0.f};
        float bias[4];
#pragma unroll
        for (int nt = 0; nt < 4; ++nt)
            bias[nt] = b_enc[n0 + wave * 64 + nt * 16 + l15];

        issue_slice_async(Wbf, wbase + 0 * WBUF_BYTES, n0, 0, tid);
        issue_slice_async(Wbf, wbase + 1 * WBUF_BYTES, n0, 1, tid);

        for (int s = 0; s < N_SLICES; ++s) {
            if (s + 2 < N_SLICES) {
                issue_slice_async(Wbf, wbase + ((s + 2) & 3) * WBUF_BYTES, n0, s + 2, tid);
                __builtin_prefetch(Wbf + (size_t)(n0 + (tid >> 2)) * D_DIM
                                   + (s + 4) * K_SLICE, 0, 0);
            }

            // A-fragments first: smA is read-only during the loop, so these
            // ds_loads overlap the asynccnt wait and the split barrier below.
            const int kabs = s * K_SLICE + kOffA;
            const int b0 = ((kabs >> 3)        ^ l15) << 3;
            const int b1 = (((kabs + 16) >> 3) ^ l15) << 3;
            union { v16bf v; v8bf h[2]; } A0, A1;
            A0.h[0] = *(const v8bf*)(smA + l15 * D_DIM + b0);
            A0.h[1] = *(const v8bf*)(smA + l15 * D_DIM + b1);
            A1.h[0] = *(const v8bf*)(smA + (16 + l15) * D_DIM + b0);
            A1.h[1] = *(const v8bf*)(smA + (16 + l15) * D_DIM + b1);

            // my slice-s copies landed -> signal; wait for everyone else.
            if (s + 2 < N_SLICES) {
                asm volatile("s_wait_asynccnt 0x10" ::: "memory");
            } else if (s + 1 < N_SLICES) {
                asm volatile("s_wait_asynccnt 0x8" ::: "memory");
            } else {
                asm volatile("s_wait_asynccnt 0x0" ::: "memory");
            }
            asm volatile("s_barrier_signal -1" ::: "memory");
            asm volatile("s_barrier_wait -1" ::: "memory");

            const __bf16* wbuf = smW + (s & 3) * (N_CHUNK * W_STRIDE);
#pragma unroll
            for (int nt = 0; nt < 4; ++nt) {
                union { v16bf v; v8bf h[2]; } B;
                const __bf16* wp = wbuf + (wave * 64 + nt * 16 + l15) * W_STRIDE + kOffB;
                B.h[0] = *(const v8bf*)(wp);
                B.h[1] = *(const v8bf*)(wp + 16);
                acc[0][nt] = __builtin_amdgcn_wmma_f32_16x16x32_bf16(
                    false, A0.v, false, B.v, (short)0, acc[0][nt], false, false);
                acc[1][nt] = __builtin_amdgcn_wmma_f32_16x16x32_bf16(
                    false, A1.v, false, B.v, (short)0, acc[1][nt], false, false);
            }
        }
        // C-stage aliases W buffers 0/1 only; computes on them finished >= 2
        // barriers ago, and in-flight computes (s=62,63) read buffers 2/3.
        chunk_epilogue(acc, bias, smC, smTV, smTI, tid, wave, lane, n0);
    }

    decode(smTV, smTI, W_dec, b_dec, out, rowBase, wave, lane);
}

// ---- fp32-W fallback (register-staged, 2 buffers) --------------------------

__global__ __launch_bounds__(256) void sae_topk_fused_fp32w(
    const float* __restrict__ x, const float* __restrict__ W_enc,
    const float* __restrict__ b_enc, const float* __restrict__ W_dec,
    const float* __restrict__ b_dec, float* __restrict__ out, int L)
{
    __shared__ __align__(16) unsigned char smem[F_TOTAL];
    __bf16* smA  = (__bf16*)(smem);
    __bf16* smW  = (__bf16*)(smem + SM_W_OFF);
    float*  smC  = (float*) (smem + SM_W_OFF);
    float*  smTV = (float*) (smem + F_TV_OFF);
    int*    smTI = (int*)   (smem + F_TI_OFF);

    const int tid  = threadIdx.x;
    const int lane = tid & 31;
    const int wave = tid >> 5;
    const int rowBase = blockIdx.x * M_WG;

    init_topk(smTV, smTI, tid);
    load_A(smA, x, b_dec, rowBase, tid);
    __syncthreads();

    const int sl_lat = tid >> 2;
    const int sl_kb  = (tid & 3) * 8;
    const int kOffA = (lane >> 4) << 3;
    const int kOffB = (lane >> 4) << 3;
    const int l15   = lane & 15;

    for (int n0 = 0; n0 < L; n0 += N_CHUNK) {
        v8f acc[2][4];
#pragma unroll
        for (int m = 0; m < 2; ++m)
#pragma unroll
            for (int nt = 0; nt < 4; ++nt)
                acc[m][nt] = (v8f){0.f,0.f,0.f,0.f,0.f,0.f,0.f,0.f};
        float bias[4];
#pragma unroll
        for (int nt = 0; nt < 4; ++nt)
            bias[nt] = b_enc[n0 + wave * 64 + nt * 16 + l15];

#pragma unroll
        for (int p = 0; p < 8; ++p) {
            const int lat = p * 64 + sl_lat;
            const float* src = W_enc + (size_t)(n0 + lat) * D_DIM + sl_kb;
            v4f f0 = *(const v4f*)(src);
            v4f f1 = *(const v4f*)(src + 4);
            *(v8bf*)(smW + lat * W_STRIDE + sl_kb) = cvt8(f0, f1);
        }
        __syncthreads();

        for (int s = 0; s < N_SLICES; ++s) {
            v4f f[8][2];
            const bool more = (s + 1 < N_SLICES);
            if (more) {
#pragma unroll
                for (int p = 0; p < 8; ++p) {
                    const float* src = W_enc + (size_t)(n0 + p * 64 + sl_lat) * D_DIM
                                       + (s + 1) * K_SLICE + sl_kb;
                    f[p][0] = *(const v4f*)(src);
                    f[p][1] = *(const v4f*)(src + 4);
                }
                __builtin_prefetch(W_enc + (size_t)(n0 + sl_lat) * D_DIM
                                   + (s + 2) * K_SLICE + sl_kb, 0, 0);
            }

            const int kabs = s * K_SLICE + kOffA;
            const int b0 = ((kabs >> 3)        ^ l15) << 3;
            const int b1 = (((kabs + 16) >> 3) ^ l15) << 3;
            union { v16bf v; v8bf h[2]; } A0, A1;
            A0.h[0] = *(const v8bf*)(smA + l15 * D_DIM + b0);
            A0.h[1] = *(const v8bf*)(smA + l15 * D_DIM + b1);
            A1.h[0] = *(const v8bf*)(smA + (16 + l15) * D_DIM + b0);
            A1.h[1] = *(const v8bf*)(smA + (16 + l15) * D_DIM + b1);

            const __bf16* wbuf = smW + (s & 1) * (N_CHUNK * W_STRIDE);
#pragma unroll
            for (int nt = 0; nt < 4; ++nt) {
                union { v16bf v; v8bf h[2]; } B;
                const __bf16* wp = wbuf + (wave * 64 + nt * 16 + l15) * W_STRIDE + kOffB;
                B.h[0] = *(const v8bf*)(wp);
                B.h[1] = *(const v8bf*)(wp + 16);
                acc[0][nt] = __builtin_amdgcn_wmma_f32_16x16x32_bf16(
                    false, A0.v, false, B.v, (short)0, acc[0][nt], false, false);
                acc[1][nt] = __builtin_amdgcn_wmma_f32_16x16x32_bf16(
                    false, A1.v, false, B.v, (short)0, acc[1][nt], false, false);
            }

            if (more) {
                __bf16* dbuf = smW + ((s + 1) & 1) * (N_CHUNK * W_STRIDE);
#pragma unroll
                for (int p = 0; p < 8; ++p)
                    *(v8bf*)(dbuf + (p * 64 + sl_lat) * W_STRIDE + sl_kb) =
                        cvt8(f[p][0], f[p][1]);
            }
            __syncthreads();
        }

        chunk_epilogue(acc, bias, smC, smTV, smTI, tid, wave, lane, n0);
    }

    decode(smTV, smTI, W_dec, b_dec, out, rowBase, wave, lane);
}

// ---------------------------------------------------------------------------

extern "C" void kernel_launch(void* const* d_in, const int* in_sizes, int n_in,
                              void* d_out, int out_size, void* d_ws, size_t ws_size,
                              hipStream_t stream) {
    (void)n_in; (void)out_size;
    const float* x     = (const float*)d_in[0];
    const float* W_enc = (const float*)d_in[1];
    const float* b_enc = (const float*)d_in[2];
    const float* W_dec = (const float*)d_in[3];
    const float* b_dec = (const float*)d_in[4];
    float* out = (float*)d_out;

    const int D = in_sizes[4];        // 2048 (b_dec length)
    const int L = in_sizes[2];        // 32768 (b_enc length)
    const int N = in_sizes[0] / D;    // 4096 tokens
    const long long wElems = (long long)L * D;

    dim3 grid(N / M_WG);              // 128 workgroups, 256 threads (8 waves)
    if (d_ws != nullptr && ws_size >= (size_t)wElems * 2) {
        __bf16* Wbf = (__bf16*)d_ws;
        convert_w_bf16<<<4096, 256, 0, stream>>>(W_enc, Wbf, wElems);
        sae_topk_fused_bf16w<<<grid, 256, 0, stream>>>(x, Wbf, b_enc, W_dec,
                                                       b_dec, out, L);
    } else {
        sae_topk_fused_fp32w<<<grid, 256, 0, stream>>>(x, W_enc, b_enc, W_dec,
                                                       b_dec, out, L);
    }
}
